// MultiHeadAttention_9620726743345
// MI455X (gfx1250) — matrix operand order repaired and checked by reading, not compiled, not run on hardware
//
#include <hip/hip_runtime.h>

// ---------------------------------------------------------------------------
// MI455X / gfx1250: fused QKV projection + 16-head flash attention.
// - All matrix math on v_wmma_f32_16x16x32_bf16 (wave32 WMMA, f32 accum).
// - V produced transposed ([b][h][dim][s]) for contiguous P@V B-fragments.
// - flash_attn stages K/V tiles in LDS via GLOBAL_LOAD_ASYNC_TO_LDS_B128
//   (ASYNCcnt-tracked, double-buffered) so the 4 waves of a block share one
//   fetch of each tile and loads overlap WMMA compute.
// ---------------------------------------------------------------------------

typedef __attribute__((ext_vector_type(8)))  float  f32x8;
typedef __attribute__((ext_vector_type(4)))  float  f32x4;
typedef __attribute__((ext_vector_type(16))) __bf16 bf16x16;
typedef __attribute__((ext_vector_type(8)))  __bf16 bf16x8;

#define BATCH    2
#define S_LEN    2048
#define D_MODEL  1024
#define N_HEADS  16
#define HEAD_DIM 64
#define TOKENS   (BATCH * S_LEN)   // 4096
#define N_QKV    (3 * D_MODEL)     // 3072

#define KPAD 72   // 64 data + 8 pad (keeps 16B align, spreads LDS banks)
#define VPAD 40   // 32 data + 8 pad

__device__ __forceinline__ f32x8 wmma_bf16(bf16x16 a, bf16x16 b, f32x8 c) {
    return __builtin_amdgcn_wmma_f32_16x16x32_bf16(
        /*neg_a=*/false, a, /*neg_b=*/false, b,
        /*c_mod=*/(short)0, c, /*reuse_a=*/false, /*reuse_b=*/false);
}

// Reduce across each 16-lane half of the wave32 (masks 1,2,4,8 keep bit4).
__device__ __forceinline__ float red_max16(float v) {
    v = fmaxf(v, __shfl_xor(v, 1, 32));
    v = fmaxf(v, __shfl_xor(v, 2, 32));
    v = fmaxf(v, __shfl_xor(v, 4, 32));
    v = fmaxf(v, __shfl_xor(v, 8, 32));
    return v;
}
__device__ __forceinline__ float red_sum16(float v) {
    v += __shfl_xor(v, 1, 32);
    v += __shfl_xor(v, 2, 32);
    v += __shfl_xor(v, 4, 32);
    v += __shfl_xor(v, 8, 32);
    return v;
}

// Async DMA: 16 bytes global -> LDS, no VGPR round trip (ASYNCcnt).
__device__ __forceinline__ void async_ld16(const __bf16* gptr, __bf16* lptr) {
    unsigned loff = (unsigned)(unsigned long long)lptr;       // LDS byte offset
    unsigned long long g = (unsigned long long)gptr;
    asm volatile("global_load_async_to_lds_b128 %0, %1, off"
                 :: "v"(loff), "v"(g) : "memory");
}

// ---------------------------------------------------------------------------
// Kernel 0: fp32 -> bf16 conversion (vectorized).
// ---------------------------------------------------------------------------
__global__ void cvt_f32_bf16(const float* __restrict__ in,
                             __bf16* __restrict__ out, int n) {
    int i = (blockIdx.x * blockDim.x + threadIdx.x) * 4;
    if (i + 3 < n) {
        f32x4 v = *(const f32x4*)(in + i);
        out[i + 0] = (__bf16)v.x;
        out[i + 1] = (__bf16)v.y;
        out[i + 2] = (__bf16)v.z;
        out[i + 3] = (__bf16)v.w;
    } else {
        for (int j = i; j < n; ++j) out[j] = (__bf16)in[j];
    }
}

// ---------------------------------------------------------------------------
// Kernel 1: qkv = x @ W^T + b.
//   Q, K written [b][h][s][64] bf16; V written TRANSPOSED [b][h][64][s] bf16.
// One wave computes a 16(M) x 64(N) tile; K = 1024 in steps of 32.
// ---------------------------------------------------------------------------
__global__ __launch_bounds__(256)
void qkv_gemm(const __bf16* __restrict__ xb, const __bf16* __restrict__ wb,
              const float* __restrict__ bias,
              __bf16* __restrict__ qb, __bf16* __restrict__ kb,
              __bf16* __restrict__ vb) {
    const int lane = threadIdx.x & 31;
    const int wave = threadIdx.x >> 5;
    const int id   = blockIdx.x * 8 + wave;     // 0..12287
    const int mt   = id / 48;                   // 16-row token tile, 0..255
    const int ntg  = id % 48;                   // 64-col group, 0..47
    const int h     = ntg / 3;
    const int which = ntg % 3;                  // 0=Q, 1=K, 2=V

    const int  nl    = lane & 15;
    const int  hi    = lane >> 4;
    const int  lo    = hi * 8;
    const int  kb16  = hi * 16;
    const int  mBase = mt * 16;
    const long aRow  = mBase + nl;
    const long nBase = (long)ntg * 64;          // == h*192 + which*64

    f32x8 acc[4] = {};

    for (int k0 = 0; k0 < D_MODEL; k0 += 32) {
        bf16x16 a;
        {
            const __bf16* ap = xb + aRow * D_MODEL + k0;
            bf16x8 a0 = *(const bf16x8*)(ap + lo);
            bf16x8 a1 = *(const bf16x8*)(ap + 16 + lo);
#pragma unroll
            for (int j = 0; j < 8; ++j) { a[j] = a0[j]; a[8 + j] = a1[j]; }
        }
#pragma unroll
        for (int nt = 0; nt < 4; ++nt) {
            const long n = nBase + nt * 16 + nl;
            bf16x8 b0v = *(const bf16x8*)(wb + n * D_MODEL + k0 + lo);
            bf16x8 b1v = *(const bf16x8*)(wb + n * D_MODEL + k0 + 16 + lo);
            bf16x16 bf;
#pragma unroll
            for (int j = 0; j < 8; ++j) { bf[j] = b0v[j]; bf[8 + j] = b1v[j]; }
            acc[nt] = wmma_bf16(a, bf, acc[nt]);
        }
    }

    // All 16 rows of this tile are in the same batch (2048 % 16 == 0).
    const int bidx = mBase >> 11;
    const int sid0 = mBase & (S_LEN - 1);

    if (which == 2) {
        // V transposed store: lane owns dim = nt*16+nl; its 8 C-layout rows
        // are contiguous s positions -> one bf16x8 store per n-tile.
        const long hbase = (long)(bidx * N_HEADS + h) * HEAD_DIM;
#pragma unroll
        for (int nt = 0; nt < 4; ++nt) {
            const int  dim = nt * 16 + nl;
            const float bv = bias[nBase + dim];
            bf16x8 vvec;
#pragma unroll
            for (int r = 0; r < 8; ++r) vvec[r] = (__bf16)(acc[nt][r] + bv);
            *(bf16x8*)(vb + (hbase + dim) * S_LEN + sid0 + hi * 8) = vvec;
        }
    } else {
        __bf16* dst = (which == 0) ? qb : kb;
#pragma unroll
        for (int nt = 0; nt < 4; ++nt) {
            const float bv = bias[nBase + nt * 16 + nl];
#pragma unroll
            for (int r = 0; r < 8; ++r) {
                const int sidx = sid0 + r + hi * 8;
                const long off = (((long)(bidx * N_HEADS + h) * S_LEN) + sidx)
                               * HEAD_DIM + nt * 16 + nl;
                dst[off] = (__bf16)(acc[nt][r] + bv);
            }
        }
    }
}

// ---------------------------------------------------------------------------
// Kernel 2: flash attention per (b,h). Wave = 16 q-rows; block = 4 waves.
// K/V tiles (32 keys) are async-DMA'd to double-buffered LDS and shared by
// all 4 waves; per 32-key block: 4 WMMAs for scores, online softmax with
// cross-lane row reductions, P transposed C->A layout through per-wave LDS,
// 4 WMMAs for P@V. Output flat layout == [b][h][s][64] fp32.
// ---------------------------------------------------------------------------
__global__ __launch_bounds__(128)
void flash_attn(const __bf16* __restrict__ qb, const __bf16* __restrict__ kb,
                const __bf16* __restrict__ vb, float* __restrict__ out) {
    __shared__ __bf16 lds_k[2][32][KPAD];   // [buf][key][dim]
    __shared__ __bf16 lds_v[2][HEAD_DIM][VPAD]; // [buf][dim][key]
    __shared__ __bf16 lds_p[4][16][32];     // per-wave P scratch (q-row x key)

    const int tid  = threadIdx.x;            // 0..127
    const int lane = tid & 31;
    const int wave = tid >> 5;
    const int nl   = lane & 15;
    const int hi   = lane >> 4;
    const int lo   = hi * 8;
    const int kb16 = hi * 16;

    const int bh   = blockIdx.x >> 5;          // 0..31
    const int qblk = blockIdx.x & 31;          // 0..31
    const long base = (long)bh * S_LEN * HEAD_DIM;   // same for q/k/vT/out
    const int  qRow0 = qblk * 64 + wave * 16;

    // Cooperative async stage of one 32-key K tile + V tile (4 x 16B/thread).
    auto issue_block = [&](int kk, int buf) {
#pragma unroll
        for (int i = 0; i < 2; ++i) {           // K: 256 chunks of 16B
            const int id  = tid + 128 * i;
            const int row = id >> 3, c8 = (id & 7) * 8;
            async_ld16(kb + base + (long)(kk + row) * HEAD_DIM + c8,
                       &lds_k[buf][row][c8]);
        }
#pragma unroll
        for (int i = 0; i < 2; ++i) {           // V: 256 chunks of 16B
            const int id  = tid + 128 * i;
            const int dim = id >> 2, kc = (id & 3) * 8;
            async_ld16(vb + base + (long)dim * S_LEN + kk + kc,
                       &lds_v[buf][dim][kc]);
        }
    };

    // Q fragments (A layout), rows qRow0 + nl, K split 0..31 / 32..63
    bf16x16 qf[2];
    {
        const __bf16* qp = qb + base + (long)(qRow0 + nl) * HEAD_DIM;
#pragma unroll
        for (int t = 0; t < 2; ++t) {
            bf16x8 a0 = *(const bf16x8*)(qp + t * 32 + lo);
            bf16x8 a1 = *(const bf16x8*)(qp + t * 32 + 16 + lo);
#pragma unroll
            for (int j = 0; j < 8; ++j) { qf[t][j] = a0[j]; qf[t][8 + j] = a1[j]; }
        }
    }

    f32x8 acc[4] = {};
    float mrow[8], lrow[8];
#pragma unroll
    for (int r = 0; r < 8; ++r) { mrow[r] = -3.0e38f; lrow[r] = 0.0f; }

    const float scale = 0.125f;   // 1/sqrt(64)

    issue_block(0, 0);            // prologue: stage first tile

    for (int k0 = 0; k0 < S_LEN; k0 += 32) {
        const int cur = (k0 >> 5) & 1;
        const bool has_next = (k0 + 32) < S_LEN;

        if (has_next) {
            issue_block(k0 + 32, cur ^ 1);
            // 8 outstanding; async loads complete in order per wave, so
            // waiting to <=4 guarantees the current tile is resident.
            asm volatile("s_wait_asynccnt 0x4" ::: "memory");
        } else {
            asm volatile("s_wait_asynccnt 0x0" ::: "memory");
        }
        __syncthreads();

        // ---- scores: two 16-key tiles (K fragments from LDS) ----
        f32x8 s[2] = {};
#pragma unroll
        for (int t = 0; t < 2; ++t) {
            const __bf16* kp = &lds_k[cur][t * 16 + nl][0];
            bf16x16 b0, b1;
            {
                bf16x8 x0 = *(const bf16x8*)(kp + lo);
                bf16x8 x1 = *(const bf16x8*)(kp + 16 + lo);
                bf16x8 y0 = *(const bf16x8*)(kp + 32 + lo);
                bf16x8 y1 = *(const bf16x8*)(kp + 48 + lo);
#pragma unroll
                for (int j = 0; j < 8; ++j) { b0[j] = x0[j]; b0[8 + j] = x1[j]; b1[j] = y0[j]; b1[8 + j] = y1[j]; }
            }
            s[t] = wmma_bf16(qf[0], b0, s[t]);
            s[t] = wmma_bf16(qf[1], b1, s[t]);
        }

        // ---- online softmax update (per C-layout row r) ----
#pragma unroll
        for (int r = 0; r < 8; ++r) {
            const float v0 = s[0][r] * scale;
            const float v1 = s[1][r] * scale;
            const float tmax = red_max16(fmaxf(v0, v1));
            const float mnew = fmaxf(mrow[r], tmax);
            const float alpha = __expf(mrow[r] - mnew);
            const float p0 = __expf(v0 - mnew);
            const float p1 = __expf(v1 - mnew);
            lrow[r] = lrow[r] * alpha + red_sum16(p0 + p1);
            mrow[r] = mnew;
#pragma unroll
            for (int nt = 0; nt < 4; ++nt) acc[nt][r] *= alpha;
            lds_p[wave][r + hi * 8][nl]      = (__bf16)p0;
            lds_p[wave][r + hi * 8][16 + nl] = (__bf16)p1;
        }

        // DS ops are in-order within a wave; stop compiler reordering + wait.
        asm volatile("" ::: "memory");
        asm volatile("s_wait_dscnt 0x0" ::: "memory");

        // ---- P in A layout (16x32): row = nl, keys per ISA A-packing ----
        bf16x16 pf;
        {
            const __bf16* pp = &lds_p[wave][nl][0];
            bf16x8 p0v = *(const bf16x8*)(pp + lo);
            bf16x8 p1v = *(const bf16x8*)(pp + 16 + lo);
#pragma unroll
            for (int j = 0; j < 8; ++j) { pf[j] = p0v[j]; pf[8 + j] = p1v[j]; }
        }
        asm volatile("" ::: "memory");

        // ---- O += P @ V  (V fragment from LDS: dim row, two 8-key runs at lo, 16+lo) ----
#pragma unroll
        for (int nt = 0; nt < 4; ++nt) {
            const int dim = nt * 16 + nl;
            bf16x8 v0v = *(const bf16x8*)(&lds_v[cur][dim][lo]);
            bf16x8 v1v = *(const bf16x8*)(&lds_v[cur][dim][16 + lo]);
            bf16x16 vf;
#pragma unroll
            for (int j = 0; j < 8; ++j) { vf[j] = v0v[j]; vf[8 + j] = v1v[j]; }
            acc[nt] = wmma_bf16(pf, vf, acc[nt]);
        }

        __syncthreads();   // all waves done with buf `cur` before it refills
    }

    // ---- normalize + store: out flat layout is [b][h][s][64] fp32 ----
#pragma unroll
    for (int r = 0; r < 8; ++r) {
        const float inv  = 1.0f / lrow[r];
        const int   qrow = qRow0 + r + hi * 8;
        const long  ob   = base + (long)qrow * HEAD_DIM;
#pragma unroll
        for (int nt = 0; nt < 4; ++nt)
            out[ob + nt * 16 + nl] = acc[nt][r] * inv;
    }
}

// ---------------------------------------------------------------------------
extern "C" void kernel_launch(void* const* d_in, const int* in_sizes, int n_in,
                              void* d_out, int out_size, void* d_ws, size_t ws_size,
                              hipStream_t stream) {
    const float* x    = (const float*)d_in[0];   // [2,2048,1024]
    const float* w    = (const float*)d_in[1];   // [3072,1024]
    const float* bias = (const float*)d_in[2];   // [3072]
    float* out = (float*)d_out;                  // [2,2048,1024] fp32

    // Workspace layout (needs ~38 MiB):
    char* ws = (char*)d_ws;
    __bf16* xb = (__bf16*)(ws);                            //  8 MiB  x bf16
    __bf16* wb = (__bf16*)(ws + (size_t)8  * 1024 * 1024); //  6 MiB  W bf16
    __bf16* qb = (__bf16*)(ws + (size_t)14 * 1024 * 1024); //  8 MiB  Q [b][h][s][64]
    __bf16* kb = (__bf16*)(ws + (size_t)22 * 1024 * 1024); //  8 MiB  K [b][h][s][64]
    __bf16* vb = (__bf16*)(ws + (size_t)30 * 1024 * 1024); //  8 MiB  V [b][h][64][s]

    const int n_x = TOKENS * D_MODEL;   // 4,194,304
    const int n_w = N_QKV * D_MODEL;    // 3,145,728

    cvt_f32_bf16<<<(n_x + 1023) / 1024, 256, 0, stream>>>(x, xb, n_x);
    cvt_f32_bf16<<<(n_w + 1023) / 1024, 256, 0, stream>>>(w, wb, n_w);

    // 256 m-tiles x 48 n-groups = 12288 waves; 8 waves/block -> 1536 blocks
    qkv_gemm<<<1536, 256, 0, stream>>>(xb, wb, bias, qb, kb, vb);

    // 32 (b,h) x 32 q-blocks = 1024 blocks of 4 waves
    flash_attn<<<1024, 128, 0, stream>>>(qb, kb, vb, out);
}
